// Guide_60997125538267
// MI455X (gfx1250) — matrix operand order repaired and checked by reading, not compiled, not run on hardware
//
#include <hip/hip_runtime.h>
#include <hip/hip_bf16.h>

typedef __attribute__((ext_vector_type(16))) _Float16 v16h;
typedef __attribute__((ext_vector_type(8)))  float    v8f;
typedef __attribute__((ext_vector_type(8)))  unsigned int v8u;

union FragU { v8u u; v16h h; };

static __device__ __forceinline__ v8f wmma_f16(const FragU& a, const FragU& b, v8f c) {
    return __builtin_amdgcn_wmma_f32_16x16x32_f16(false, a.h, false, b.h, (short)0, c, false, false);
}

struct __align__(8) H4 { _Float16 a, b, c, d; };
static __device__ __forceinline__ void store4h(_Float16* p, float4 v) {
    H4 h{(_Float16)v.x, (_Float16)v.y, (_Float16)v.z, (_Float16)v.w};
    *(H4*)p = h;
}

#define LDSTR 40   // halves per LDS row (32 + 8 pad), keeps 4B/8B alignment

// ---------------------------------------------------------------------------
// Kernel 1: f01[b, 0:256] = msg0 @ fw0.T + fb0 ; f01[b, 256:512] = msg0 @ fw1.T + fb1
// M=4096, N=512, K=4096. Tile 128x128, K-chunk 32. 8 waves: 4(M) x 2(N).
// Double-buffered LDS: next chunk's global loads issued before WMMA block.
// ---------------------------------------------------------------------------
__global__ __launch_bounds__(256) void k_f01(const float* __restrict__ msg0,
                                             const float* __restrict__ fw0,
                                             const float* __restrict__ fb0,
                                             const float* __restrict__ fw1,
                                             const float* __restrict__ fb1,
                                             float* __restrict__ f01) {
    __shared__ _Float16 As[2][128 * LDSTR];
    __shared__ _Float16 Bs[2][128 * LDSTR];

    const int tid  = threadIdx.x;
    const int lane = tid & 31;
    const int wave = tid >> 5;
    const int wm   = wave & 3;   // 0..3 -> 2 m-subtiles each
    const int wn   = wave >> 2;  // 0..1 -> 4 n-subtiles each
    const int half = lane >> 4;
    const int l16  = lane & 15;

    const int m_base = blockIdx.x * 128;
    const int n_base = blockIdx.y * 128;
    const float* fwsrc = (n_base < 256) ? fw0 : fw1;
    const float* fbsrc = (n_base < 256) ? fb0 : fb1;
    const int    n_off = (n_base < 256) ? n_base : (n_base - 256);

    const int ldrow = tid >> 3;        // 0..31 x (q -> +32)
    const int ldc4  = tid & 7;         // float4 column within 32-k chunk

    v8f acc[2][4] = {};
    float4 av[4], bv[4];

    // prologue: stage chunk 0
#pragma unroll
    for (int q = 0; q < 4; ++q) {
        int row = ldrow + q * 32;
        av[q] = *(const float4*)(msg0  + (size_t)(m_base + row) * 4096 + ldc4 * 4);
        bv[q] = *(const float4*)(fwsrc + (size_t)(n_off  + row) * 4096 + ldc4 * 4);
    }
#pragma unroll
    for (int q = 0; q < 4; ++q) {
        int row = ldrow + q * 32;
        store4h(&As[0][row * LDSTR + ldc4 * 4], av[q]);
        store4h(&Bs[0][row * LDSTR + ldc4 * 4], bv[q]);
    }
    __syncthreads();

    for (int kt = 0; kt < 4096 / 32; ++kt) {
        const int cur = kt & 1;
        // issue next chunk's global loads early (latency overlapped with WMMA)
        if (kt < 4096 / 32 - 1) {
            const int kb = (kt + 1) * 32;
#pragma unroll
            for (int q = 0; q < 4; ++q) {
                int row = ldrow + q * 32;
                av[q] = *(const float4*)(msg0  + (size_t)(m_base + row) * 4096 + kb + ldc4 * 4);
                bv[q] = *(const float4*)(fwsrc + (size_t)(n_off  + row) * 4096 + kb + ldc4 * 4);
            }
        }

        FragU a[2], bf[4];
#pragma unroll
        for (int i = 0; i < 2; ++i) {
            int m = (wm * 2 + i) * 16 + l16;
#pragma unroll
            for (int r = 0; r < 8; ++r) {
                int k = (r < 4) ? (8 * half + 2 * r) : (16 + 8 * half + 2 * (r - 4));
                a[i].u[r] = *(const unsigned int*)&As[cur][m * LDSTR + k];
            }
        }
#pragma unroll
        for (int j = 0; j < 4; ++j) {
            int n = (wn * 4 + j) * 16 + l16;
#pragma unroll
            for (int r = 0; r < 8; ++r) {
                int k = (r < 4) ? (8 * half + 2 * r) : (16 + 8 * half + 2 * (r - 4));
                bf[j].u[r] = *(const unsigned int*)&Bs[cur][n * LDSTR + k];
            }
        }
#pragma unroll
        for (int i = 0; i < 2; ++i)
#pragma unroll
            for (int j = 0; j < 4; ++j)
                acc[i][j] = wmma_f16(a[i], bf[j], acc[i][j]);

        if (kt < 4096 / 32 - 1) {
            const int nxt = cur ^ 1;
#pragma unroll
            for (int q = 0; q < 4; ++q) {
                int row = ldrow + q * 32;
                store4h(&As[nxt][row * LDSTR + ldc4 * 4], av[q]);
                store4h(&Bs[nxt][row * LDSTR + ldc4 * 4], bv[q]);
            }
        }
        __syncthreads();
    }

#pragma unroll
    for (int i = 0; i < 2; ++i)
#pragma unroll
        for (int j = 0; j < 4; ++j) {
            int n_g  = n_base + (wn * 4 + j) * 16 + l16;
            float bias = fbsrc[(n_base < 256) ? n_g : (n_g - 256)];
#pragma unroll
            for (int r = 0; r < 8; ++r) {
                int m_g = m_base + (wm * 2 + i) * 16 + 8 * half + r;
                f01[(size_t)m_g * 512 + n_g] = acc[i][j][r] + bias;
            }
        }
}

// ---------------------------------------------------------------------------
// Kernel 2: mask = softmax(msg1 @ aw.T + ab)  -> [B, 9]
// One block (256 threads) per batch row; memory-bound on msg1.
// ---------------------------------------------------------------------------
__global__ __launch_bounds__(256) void k_mask(const float* __restrict__ msg1,
                                              const float* __restrict__ aw,
                                              const float* __restrict__ ab,
                                              float* __restrict__ maskout) {
    const int b = blockIdx.x, tid = threadIdx.x;
    float acc[9] = {};
#pragma unroll
    for (int q = 0; q < 4; ++q) {
        int k = tid * 4 + q * 1024;
        float4 m = *(const float4*)(msg1 + (size_t)b * 4096 + k);
#pragma unroll
        for (int p = 0; p < 9; ++p) {
            float4 a = *(const float4*)(aw + (size_t)p * 4096 + k);
            acc[p] += m.x * a.x + m.y * a.y + m.z * a.z + m.w * a.w;
        }
    }
    __shared__ float red[9][256];
#pragma unroll
    for (int p = 0; p < 9; ++p) red[p][tid] = acc[p];
    __syncthreads();
    for (int s = 128; s > 0; s >>= 1) {
        if (tid < s) {
#pragma unroll
            for (int p = 0; p < 9; ++p) red[p][tid] += red[p][tid + s];
        }
        __syncthreads();
    }
    if (tid == 0) {
        float lg[9], mx = -1e30f;
#pragma unroll
        for (int p = 0; p < 9; ++p) { lg[p] = red[p][0] + ab[p]; mx = fmaxf(mx, lg[p]); }
        float sum = 0.f;
#pragma unroll
        for (int p = 0; p < 9; ++p) { lg[p] = expf(lg[p] - mx); sum += lg[p]; }
        float inv = 1.f / sum;
#pragma unroll
        for (int p = 0; p < 9; ++p) maskout[(size_t)b * 9 + p] = lg[p] * inv;
    }
}

// ---------------------------------------------------------------------------
// Kernel 3: g[b, c*9+p] = sum_o f1[b,o] * conv_w[o,c,p]
// GEMM M=4096, K=256, N=2304 (conv_w reshaped [256][2304] row-major = [k][n]).
// Double-buffered; output stored as f16 to halve workspace traffic.
// ---------------------------------------------------------------------------
__global__ __launch_bounds__(256) void k_g(const float* __restrict__ f01,
                                           const float* __restrict__ convw,
                                           _Float16* __restrict__ g) {
    __shared__ _Float16 As[2][128 * LDSTR];
    __shared__ _Float16 Bs[2][128 * LDSTR];

    const int tid  = threadIdx.x;
    const int lane = tid & 31;
    const int wave = tid >> 5;
    const int wm   = wave & 3;
    const int wn   = wave >> 2;
    const int half = lane >> 4;
    const int l16  = lane & 15;

    const int m_base = blockIdx.x * 128;
    const int n_base = blockIdx.y * 128;

    const int ldrow = tid >> 3;   // A-tile row group
    const int ldc4  = tid & 7;
    const int bj    = tid >> 5;   // B-tile k row group (0..7, +8 per q? no: idx>>5)
    (void)bj;

    v8f acc[2][4] = {};
    float4 av[4], bv[4];

    auto loadA = [&](int kb) {
#pragma unroll
        for (int q = 0; q < 4; ++q) {
            int row = ldrow + q * 32;
            av[q] = *(const float4*)(f01 + (size_t)(m_base + row) * 512 + 256 + kb + ldc4 * 4);
        }
    };
    auto loadB = [&](int kb) {
#pragma unroll
        for (int q = 0; q < 4; ++q) {
            int idx = tid + q * 256;
            int j = idx >> 5, c4 = idx & 31;
            bv[q] = *(const float4*)(convw + (size_t)(kb + j) * 2304 + n_base + c4 * 4);
        }
    };
    auto storeTiles = [&](int buf) {
#pragma unroll
        for (int q = 0; q < 4; ++q) {
            int row = ldrow + q * 32;
            store4h(&As[buf][row * LDSTR + ldc4 * 4], av[q]);
        }
#pragma unroll
        for (int q = 0; q < 4; ++q) {
            int idx = tid + q * 256;
            int j = idx >> 5, n0 = (idx & 31) * 4;
            Bs[buf][(n0 + 0) * LDSTR + j] = (_Float16)bv[q].x;
            Bs[buf][(n0 + 1) * LDSTR + j] = (_Float16)bv[q].y;
            Bs[buf][(n0 + 2) * LDSTR + j] = (_Float16)bv[q].z;
            Bs[buf][(n0 + 3) * LDSTR + j] = (_Float16)bv[q].w;
        }
    };

    loadA(0); loadB(0); storeTiles(0);
    __syncthreads();

    for (int kt = 0; kt < 256 / 32; ++kt) {
        const int cur = kt & 1;
        if (kt < 256 / 32 - 1) { loadA((kt + 1) * 32); loadB((kt + 1) * 32); }

        FragU a[2], bf[4];
#pragma unroll
        for (int i = 0; i < 2; ++i) {
            int m = (wm * 2 + i) * 16 + l16;
#pragma unroll
            for (int r = 0; r < 8; ++r) {
                int k = (r < 4) ? (8 * half + 2 * r) : (16 + 8 * half + 2 * (r - 4));
                a[i].u[r] = *(const unsigned int*)&As[cur][m * LDSTR + k];
            }
        }
#pragma unroll
        for (int j = 0; j < 4; ++j) {
            int n = (wn * 4 + j) * 16 + l16;
#pragma unroll
            for (int r = 0; r < 8; ++r) {
                int k = (r < 4) ? (8 * half + 2 * r) : (16 + 8 * half + 2 * (r - 4));
                bf[j].u[r] = *(const unsigned int*)&Bs[cur][n * LDSTR + k];
            }
        }
#pragma unroll
        for (int i = 0; i < 2; ++i)
#pragma unroll
            for (int j = 0; j < 4; ++j)
                acc[i][j] = wmma_f16(a[i], bf[j], acc[i][j]);

        if (kt < 256 / 32 - 1) storeTiles(cur ^ 1);
        __syncthreads();
    }

#pragma unroll
    for (int i = 0; i < 2; ++i)
#pragma unroll
        for (int j = 0; j < 4; ++j) {
            int n_g = n_base + (wn * 4 + j) * 16 + l16;
#pragma unroll
            for (int r = 0; r < 8; ++r) {
                int m_g = m_base + (wm * 2 + i) * 16 + 8 * half + r;
                g[(size_t)m_g * 2304 + n_g] = (_Float16)acc[i][j][r];
            }
        }
}

// ---------------------------------------------------------------------------
// Kernel 4: per-sample epilogue.
//   l_emb[n][c] = sum_j emb[landmarks[b,n,j]][c]
//   term1[n]    = sum_c l_emb[n][c] * f0[c]
//   s[p][n]     = sum_c l_emb[n][c] * g[b][c*9+p]
//   logits[n]   = term1[n] + sum_p mask[p] * s[p][shift(n,p)]   (3x3, zero pad)
//   out[b]      = softmax(logits)
// ---------------------------------------------------------------------------
__global__ __launch_bounds__(256) void k_final(const float* __restrict__ f01,
                                               const float* __restrict__ maskws,
                                               const _Float16* __restrict__ g,
                                               const int* __restrict__ landmarks,
                                               const float* __restrict__ emb,
                                               float* __restrict__ out) {
    const int b = blockIdx.x, tid = threadIdx.x;
    __shared__ float le[16 * 257];   // stride 257: consecutive n -> consecutive banks
    __shared__ float gbuf[2304];
    __shared__ float f0s[256];
    __shared__ float msk[9];
    __shared__ float tpart[256];
    __shared__ float term1[16];
    __shared__ float sv[9][16];
    __shared__ float logits[16];
    __shared__ float inv_sum;

    f0s[tid] = f01[(size_t)b * 512 + tid];
    if (tid < 9) msk[tid] = maskws[(size_t)b * 9 + tid];
    for (int i = tid; i < 2304; i += 256) gbuf[i] = (float)g[(size_t)b * 2304 + i];

    const int n  = tid >> 4;
    const int ci = (tid & 15) * 16;
    float vv[16];
#pragma unroll
    for (int cc = 0; cc < 16; ++cc) vv[cc] = 0.f;
    const int* lmp = landmarks + (size_t)b * 96 + n * 6;
#pragma unroll
    for (int j = 0; j < 6; ++j) {
        const float* row = emb + (size_t)lmp[j] * 256 + ci;
#pragma unroll
        for (int cc = 0; cc < 16; ++cc) vv[cc] += row[cc];
    }
#pragma unroll
    for (int cc = 0; cc < 16; ++cc) le[n * 257 + ci + cc] = vv[cc];
    __syncthreads();

    float p1 = 0.f;
#pragma unroll
    for (int cc = 0; cc < 16; ++cc) p1 += le[n * 257 + ci + cc] * f0s[ci + cc];
    tpart[tid] = p1;
    __syncthreads();

    if (tid < 16) {
        float s = 0.f;
#pragma unroll
        for (int q = 0; q < 16; ++q) s += tpart[tid * 16 + q];
        term1[tid] = s;
    }
    if (tid < 144) {
        int p = tid / 16, nn = tid % 16;
        float s = 0.f;
        for (int c = 0; c < 256; ++c) s += le[nn * 257 + c] * gbuf[c * 9 + p];
        sv[p][nn] = s;
    }
    __syncthreads();

    if (tid < 16) {
        int h = tid >> 2, w = tid & 3;
        float acc = term1[tid];
#pragma unroll
        for (int di = 0; di < 3; ++di)
#pragma unroll
            for (int dj = 0; dj < 3; ++dj) {
                int hh = h + di - 1, ww = w + dj - 1;
                if (hh >= 0 && hh < 4 && ww >= 0 && ww < 4)
                    acc += msk[di * 3 + dj] * sv[di * 3 + dj][hh * 4 + ww];
            }
        logits[tid] = acc;
    }
    __syncthreads();
    if (tid == 0) {
        float mx = logits[0];
        for (int i = 1; i < 16; ++i) mx = fmaxf(mx, logits[i]);
        float sum = 0.f;
        for (int i = 0; i < 16; ++i) { float e = expf(logits[i] - mx); logits[i] = e; sum += e; }
        inv_sum = 1.f / sum;
    }
    __syncthreads();
    if (tid < 16) out[(size_t)b * 16 + tid] = logits[tid] * inv_sum;
}

// ---------------------------------------------------------------------------
extern "C" void kernel_launch(void* const* d_in, const int* in_sizes, int n_in,
                              void* d_out, int out_size, void* d_ws, size_t ws_size,
                              hipStream_t stream) {
    (void)in_sizes; (void)n_in; (void)out_size; (void)ws_size;
    const float* msg0      = (const float*)d_in[0];
    const float* msg1      = (const float*)d_in[1];
    const int*   landmarks = (const int*)  d_in[2];
    const float* emb       = (const float*)d_in[3];
    const float* fw0       = (const float*)d_in[4];
    const float* fb0       = (const float*)d_in[5];
    const float* fw1       = (const float*)d_in[6];
    const float* fb1       = (const float*)d_in[7];
    const float* aw        = (const float*)d_in[8];
    const float* ab        = (const float*)d_in[9];
    const float* convw     = (const float*)d_in[10];
    float* out = (float*)d_out;

    // workspace layout: f01 [4096*512] f32 | mask [4096*16] f32 (padded) | g [4096*2304] f16
    float*    f01    = (float*)d_ws;
    float*    maskws = f01 + (size_t)4096 * 512;
    _Float16* g      = (_Float16*)(maskws + (size_t)4096 * 16);

    k_f01  <<<dim3(32, 4),  256, 0, stream>>>(msg0, fw0, fb0, fw1, fb1, f01);
    k_mask <<<4096,         256, 0, stream>>>(msg1, aw, ab, maskws);
    k_g    <<<dim3(32, 18), 256, 0, stream>>>(f01, convw, g);
    k_final<<<4096,         256, 0, stream>>>(f01, maskws, g, landmarks, emb, out);
}
